// sLSTMNetwork_46840913330314
// MI455X (gfx1250) — matrix operand order repaired and checked by reading, not compiled, not run on hardware
//
#include <hip/hip_runtime.h>

#define B_   64
#define S_   512
#define I_   64
#define H_   512
#define G_   2048
#define O_   24
#define NBLK 64
#define NTHR 256

typedef __attribute__((ext_vector_type(16))) __bf16 v16bf;
typedef __attribute__((ext_vector_type(8)))  __bf16 v8bf;
typedef __attribute__((ext_vector_type(8)))  float  v8f;

struct SParams {
    const float *x, *Wx0, *bx0, *Wh0, *bh0, *Wx1, *bx1, *Wh1, *bh1;
    const float *g_in, *b_in, *g_hu, *b_hu, *g_c, *b_c, *g_h, *b_h;
    const float *Wfc, *bfc;
    float* out;
    unsigned* cnt;
    unsigned short *h0_bf, *h1_bf, *x_bf, *wx0_bf, *wh0_bf, *wx1_bf, *wh1_bf;
    float *u_x0, *u_h0, *u_x1, *u_h1;
};

// float -> bf16, round-to-nearest-even
__device__ __forceinline__ unsigned short f2bf(float f) {
    unsigned u = __float_as_uint(f);
    u += 0x7fffu + ((u >> 16) & 1u);
    return (unsigned short)(u >> 16);
}

// Device-wide barrier: monotone counter, 64 blocks, reset by init kernel each call.
__device__ __forceinline__ void grid_sync(unsigned* cnt, unsigned* phase) {
    __syncthreads();
    if (threadIdx.x == 0) {
        __threadfence();
        unsigned tgt = ++(*phase) * (unsigned)NBLK;
        atomicAdd(cnt, 1u);
        while (*((volatile unsigned*)cnt) < tgt) { __builtin_amdgcn_s_sleep(1); }
        __threadfence();
    }
    __syncthreads();
}

__device__ __forceinline__ float blk_sum(float v, float* s) {
#pragma unroll
    for (int o = 16; o > 0; o >>= 1) v += __shfl_down(v, o, 32);
    __syncthreads();
    if ((threadIdx.x & 31) == 0) s[threadIdx.x >> 5] = v;
    __syncthreads();
    return s[0] + s[1] + s[2] + s[3] + s[4] + s[5] + s[6] + s[7];
}

__device__ __forceinline__ float blk_max(float v, float* s) {
#pragma unroll
    for (int o = 16; o > 0; o >>= 1) v = fmaxf(v, __shfl_down(v, o, 32));
    __syncthreads();
    if ((threadIdx.x & 31) == 0) s[threadIdx.x >> 5] = v;
    __syncthreads();
    return fmaxf(fmaxf(fmaxf(s[0], s[1]), fmaxf(s[2], s[3])),
                 fmaxf(fmaxf(s[4], s[5]), fmaxf(s[6], s[7])));
}

// A-matrix 16x32 bf16 fragment: lane holds row (lane&15); elems 0..7 = K kb..kb+7,
// elems 8..15 = K kb+16..kb+23, kb = 8*(lane>=16). Two contiguous 16B loads.
__device__ __forceinline__ v16bf ldfragA(const unsigned short* p) {
    v8bf lo = *(const v8bf*)(p);
    v8bf hi = *(const v8bf*)(p + 16);
    return __builtin_shufflevector(lo, hi, 0, 1, 2, 3, 4, 5, 6, 7,
                                   8, 9, 10, 11, 12, 13, 14, 15);
}
// B-matrix 32x16 bf16: lane holds column (lane&15); same K order as A fragment.
// B = W^T so a B column is a contiguous weight row chunk: two 16B loads.
__device__ __forceinline__ v16bf ldfragB(const unsigned short* p) {
    v8bf lo = *(const v8bf*)(p);
    v8bf hi = *(const v8bf*)(p + 16);
    return __builtin_shufflevector(lo, hi, 0, 1, 2, 3, 4, 5, 6, 7,
                                   8, 9, 10, 11, 12, 13, 14, 15);
}

// One 16x16 f32 tile of  out[m0..m0+15, n0..n0+15] = A[m0.., :K] * W[n0.., :K]^T
// NC = independent split-K accumulator chains (breaks the WMMA->WMMA RAW chain;
// with only 2 waves/SIMD resident, multiple chains are needed to keep XDL busy).
template <int NC>
__device__ __forceinline__ void gemm_tile(float* __restrict__ out, int ldo,
                                          const unsigned short* __restrict__ A, int lda,
                                          const unsigned short* __restrict__ Bw, int ldb,
                                          int m0, int n0, int K) {
    const int lane = threadIdx.x & 31;
    const int r    = lane & 15;
    const int hi   = lane >> 4;
    const unsigned short* ap = A  + (m0 + r) * lda + hi * 8;
    const unsigned short* bp = Bw + (n0 + r) * ldb + hi * 8;

    v8f acc[NC];
#pragma unroll
    for (int c = 0; c < NC; ++c) acc[c] = (v8f){};

    const int Kc = K / NC;            // K-chunk per chain (multiple of 32)
    for (int k = 0; k < Kc; k += 32) {
#pragma unroll
        for (int c = 0; c < NC; ++c) {
            const int kk = k + c * Kc;
            __builtin_prefetch(bp + kk + 256, 0, 1);
            v16bf a = ldfragA(ap + kk);
            v16bf b = ldfragB(bp + kk);
            acc[c] = __builtin_amdgcn_wmma_f32_16x16x32_bf16(false, a, false, b,
                                                             (short)0, acc[c],
                                                             false, false);
        }
    }
    v8f tot = acc[0];
#pragma unroll
    for (int c = 1; c < NC; ++c) tot += acc[c];

    // C/D layout: VGPR i -> row i + 8*(lane>=16), col = lane&15
    float* op = out + (m0 + hi * 8) * ldo + (n0 + r);
#pragma unroll
    for (int i = 0; i < 8; ++i) op[i * ldo] = tot[i];
}

// Full sLSTM cell for batch row b (block-local): dual LN over G, gates,
// exp-input-gate, max-abs stabilizer, LN(c), LN(h). c-state lives in LDS.
__device__ __forceinline__ void cell_update(
    int b, int do_fc,
    const float* __restrict__ ux, const float* __restrict__ uh,
    const float* __restrict__ bx, const float* __restrict__ bh,
    const float* __restrict__ gin, const float* __restrict__ bin,
    const float* __restrict__ ghu, const float* __restrict__ bhu,
    const float* __restrict__ gc,  const float* __restrict__ bc,
    const float* __restrict__ ghn, const float* __restrict__ bhn,
    float* cstate, unsigned short* __restrict__ hbf,
    float* red, float* hrow,
    const float* __restrict__ Wfc, const float* __restrict__ bfc,
    float* __restrict__ out)
{
    const int tid = threadIdx.x;
    const float* uxr = ux + b * G_;
    const float* uhr = uh + b * G_;

    float sx = 0.f, sxx = 0.f, sh = 0.f, shh = 0.f;
    for (int n = tid; n < G_; n += NTHR) {
        float vx = uxr[n] + bx[n];
        float vh = uhr[n] + bh[n];
        sx += vx; sxx += vx * vx; sh += vh; shh += vh * vh;
    }
    sx  = blk_sum(sx, red);  sxx = blk_sum(sxx, red);
    sh  = blk_sum(sh, red);  shh = blk_sum(shh, red);
    const float inv_g = 1.0f / (float)G_;
    float mux = sx * inv_g, muh = sh * inv_g;
    float rsx = rsqrtf(fmaxf(sxx * inv_g - mux * mux, 0.f) + 1e-5f);
    float rsh = rsqrtf(fmaxf(shh * inv_g - muh * muh, 0.f) + 1e-5f);

    auto pre = [&](int n) -> float {
        float a = (uxr[n] + bx[n] - mux) * rsx * gin[n] + bin[n];
        float c = (uhr[n] + bh[n] - muh) * rsh * ghu[n] + bhu[n];
        return a + c;
    };

    float cv[2], ov[2];
#pragma unroll
    for (int j = 0; j < 2; ++j) {
        int h = tid + j * NTHR;
        float si = pre(h);
        float sf = pre(h + H_);
        float sg = pre(h + 2 * H_);
        float so = pre(h + 3 * H_);
        float iv = expf(si);                      // exponential input gate
        float fv = 1.f / (1.f + expf(-sf));
        float gv = tanhf(sg);
        ov[j] = 1.f / (1.f + expf(-so));
        cv[j] = fv * cstate[h] + iv * gv;
    }

    // per-row max-abs stabilizer
    float mx = blk_max(fmaxf(fabsf(cv[0]), fabsf(cv[1])), red);
    float sc = 1.f / (mx + 1e-6f);
    cv[0] *= sc; cv[1] *= sc;

    // LN over cell state (512)
    float s1 = blk_sum(cv[0] + cv[1], red);
    float s2 = blk_sum(cv[0] * cv[0] + cv[1] * cv[1], red);
    const float inv_h = 1.0f / (float)H_;
    float muc = s1 * inv_h;
    float rsc = rsqrtf(fmaxf(s2 * inv_h - muc * muc, 0.f) + 1e-5f);

    float hv[2];
#pragma unroll
    for (int j = 0; j < 2; ++j) {
        int h = tid + j * NTHR;
        float cl = (cv[j] - muc) * rsc * gc[h] + bc[h];
        cstate[h] = cl;                           // post-LN c carried forward
        hv[j] = ov[j] * tanhf(cl);
    }

    // LN over hidden (512)
    float t1 = blk_sum(hv[0] + hv[1], red);
    float t2 = blk_sum(hv[0] * hv[0] + hv[1] * hv[1], red);
    float muo = t1 * inv_h;
    float rso = rsqrtf(fmaxf(t2 * inv_h - muo * muo, 0.f) + 1e-5f);
#pragma unroll
    for (int j = 0; j < 2; ++j) {
        int h = tid + j * NTHR;
        float hl = (hv[j] - muo) * rso * ghn[h] + bhn[h];
        hbf[b * H_ + h] = f2bf(hl);
        if (do_fc) hrow[h] = hl;
    }

    if (do_fc) {                                  // final fc head, last timestep only
        __syncthreads();
        if (tid < O_) {
            float acc = bfc[tid];
            const float* wr = Wfc + tid * H_;
            for (int k = 0; k < H_; ++k) acc += hrow[k] * wr[k];
            out[b * O_ + tid] = acc;
        }
    }
}

__global__ void slstm_init(unsigned* cnt) {
    if (threadIdx.x == 0) *cnt = 0u;
}

__global__ __launch_bounds__(NTHR) void slstm_persistent(SParams p) {
    __shared__ float cst0[H_];
    __shared__ float cst1[H_];
    __shared__ float hrow[H_];
    __shared__ float red[8];

    const int tid = threadIdx.x;
    const int blk = blockIdx.x;
    unsigned phase = 0;

    // ---- one-time convert fp32 -> bf16 (weights, x) + zero states ----
    const int gtid = blk * NTHR + tid;
    const int gsz  = NBLK * NTHR;
    for (int i = gtid; i < G_ * I_; i += gsz) p.wx0_bf[i] = f2bf(p.Wx0[i]);
    for (int i = gtid; i < G_ * H_; i += gsz) {
        p.wh0_bf[i] = f2bf(p.Wh0[i]);
        p.wx1_bf[i] = f2bf(p.Wx1[i]);
        p.wh1_bf[i] = f2bf(p.Wh1[i]);
    }
    for (int i = gtid; i < B_ * S_ * I_; i += gsz) p.x_bf[i] = f2bf(p.x[i]);
    for (int i = gtid; i < B_ * H_; i += gsz) { p.h0_bf[i] = 0; p.h1_bf[i] = 0; }
    cst0[tid] = 0.f; cst0[tid + NTHR] = 0.f;
    cst1[tid] = 0.f; cst1[tid + NTHR] = 0.f;
    grid_sync(p.cnt, &phase);

    // one 16x16 output tile per wave: 64 blocks * 8 waves = 512 tiles = (64/16)*(2048/16)
    const int wave = tid >> 5;
    const int gt = blk * 8 + wave;
    const int m0 = (gt >> 7) * 16;
    const int n0 = (gt & 127) * 16;

    for (int t = 0; t < S_; ++t) {
        // Phase A: u_h0 = h0*Wh0^T (K=512, 4 split-K chains), u_x0 = x_t*Wx0^T (K=64)
        gemm_tile<4>(p.u_h0, G_, p.h0_bf, H_, p.wh0_bf, H_, m0, n0, H_);
        gemm_tile<2>(p.u_x0, G_, p.x_bf + t * I_, S_ * I_, p.wx0_bf, I_, m0, n0, I_);
        grid_sync(p.cnt, &phase);

        // Phase B: layer-0 cell, block b owns batch row b
        cell_update(blk, 0, p.u_x0, p.u_h0, p.bx0, p.bh0,
                    p.g_in, p.b_in, p.g_hu, p.b_hu,
                    p.g_c, p.b_c, p.g_h, p.b_h,
                    cst0, p.h0_bf, red, hrow, nullptr, nullptr, nullptr);
        grid_sync(p.cnt, &phase);

        // Phase C: u_x1 = h0*Wx1^T, u_h1 = h1*Wh1^T (both K=512)
        gemm_tile<4>(p.u_x1, G_, p.h0_bf, H_, p.wx1_bf, H_, m0, n0, H_);
        gemm_tile<4>(p.u_h1, G_, p.h1_bf, H_, p.wh1_bf, H_, m0, n0, H_);
        grid_sync(p.cnt, &phase);

        // Phase D: layer-1 cell (+ fc head on last step)
        cell_update(blk, (t == S_ - 1) ? 1 : 0, p.u_x1, p.u_h1, p.bx1, p.bh1,
                    p.g_in + G_, p.b_in + G_, p.g_hu + G_, p.b_hu + G_,
                    p.g_c + H_, p.b_c + H_, p.g_h + H_, p.b_h + H_,
                    cst1, p.h1_bf, red, hrow, p.Wfc, p.bfc, p.out);
        grid_sync(p.cnt, &phase);
    }
}

extern "C" void kernel_launch(void* const* d_in, const int* in_sizes, int n_in,
                              void* d_out, int out_size, void* d_ws, size_t ws_size,
                              hipStream_t stream) {
    (void)in_sizes; (void)n_in; (void)out_size; (void)ws_size;

    SParams p;
    p.x    = (const float*)d_in[0];
    p.Wx0  = (const float*)d_in[1];
    p.bx0  = (const float*)d_in[2];
    p.Wh0  = (const float*)d_in[3];
    p.bh0  = (const float*)d_in[4];
    p.Wx1  = (const float*)d_in[5];
    p.bx1  = (const float*)d_in[6];
    p.Wh1  = (const float*)d_in[7];
    p.bh1  = (const float*)d_in[8];
    p.g_in = (const float*)d_in[9];
    p.b_in = (const float*)d_in[10];
    p.g_hu = (const float*)d_in[11];
    p.b_hu = (const float*)d_in[12];
    p.g_c  = (const float*)d_in[13];
    p.b_c  = (const float*)d_in[14];
    p.g_h  = (const float*)d_in[15];
    p.b_h  = (const float*)d_in[16];
    p.Wfc  = (const float*)d_in[17];
    p.bfc  = (const float*)d_in[18];
    p.out  = (float*)d_out;

    char* ws = (char*)d_ws;
    size_t off = 0;
    p.cnt = (unsigned*)(ws + off);            off += 256;
    p.h0_bf  = (unsigned short*)(ws + off);   off += (size_t)B_ * H_ * 2;
    p.h1_bf  = (unsigned short*)(ws + off);   off += (size_t)B_ * H_ * 2;
    p.x_bf   = (unsigned short*)(ws + off);   off += (size_t)B_ * S_ * I_ * 2;
    p.wx0_bf = (unsigned short*)(ws + off);   off += (size_t)G_ * I_ * 2;
    p.wh0_bf = (unsigned short*)(ws + off);   off += (size_t)G_ * H_ * 2;
    p.wx1_bf = (unsigned short*)(ws + off);   off += (size_t)G_ * H_ * 2;
    p.wh1_bf = (unsigned short*)(ws + off);   off += (size_t)G_ * H_ * 2;
    p.u_x0 = (float*)(ws + off);              off += (size_t)B_ * G_ * 4;
    p.u_h0 = (float*)(ws + off);              off += (size_t)B_ * G_ * 4;
    p.u_x1 = (float*)(ws + off);              off += (size_t)B_ * G_ * 4;
    p.u_h1 = (float*)(ws + off);              off += (size_t)B_ * G_ * 4;

    slstm_init<<<1, 32, 0, stream>>>(p.cnt);
    slstm_persistent<<<NBLK, NTHR, 0, stream>>>(p);
}
